// TextAttn_43911745634336
// MI455X (gfx1250) — matrix operand order repaired and checked by reading, not compiled, not run on hardware
//
#include <hip/hip_runtime.h>
#include <hip/hip_bf16.h>

// ---- sizes from the reference ----
#define B_DIM  16
#define NL_DIM 4096
#define NG_DIM 1024
#define D_DIM  256

typedef __attribute__((ext_vector_type(16))) __bf16 v16bf;
typedef __attribute__((ext_vector_type(8)))  __bf16 v8bf;
typedef __attribute__((ext_vector_type(8)))  float  v8f;

// One per-lane 16B async global->LDS copy (GLOBAL_LOAD_ASYNC_TO_LDS_B128,
// GV addressing: VDST = LDS byte offset VGPR, VADDR = 64-bit global address).
#define ASYNC_B128(ldsOff, gaddr)                                              \
  asm volatile("global_load_async_to_lds_b128 %0, %1, off"                     \
               :: "v"(ldsOff), "v"(gaddr) : "memory")

#define WAIT_ASYNC() asm volatile("s_wait_asynccnt 0x0" ::: "memory")
#define LDS_FENCE()  asm volatile("" ::: "memory")

// ---------------------------------------------------------------------------
// Fragment builders (CDNA5 wave32 WMMA bf16 layouts, ISA 7.12.2)
// ---------------------------------------------------------------------------
static __device__ __forceinline__ void cvt4(v16bf& r, int o, float4 v) {
  r[o + 0] = (__bf16)v.x; r[o + 1] = (__bf16)v.y;
  r[o + 2] = (__bf16)v.z; r[o + 3] = (__bf16)v.w;
}

// elems 0..7 <- f32 p[0..7]; elems 8..15 <- f32 p[16..23]
static __device__ __forceinline__ v16bf frag_f32_split(const float* __restrict__ p) {
  v16bf r;
  cvt4(r, 0,  *(const float4*)(p + 0));
  cvt4(r, 4,  *(const float4*)(p + 4));
  cvt4(r, 8,  *(const float4*)(p + 16));
  cvt4(r, 12, *(const float4*)(p + 20));
  return r;
}

// elems 0..15 <- f32 p[0..15]
static __device__ __forceinline__ v16bf frag_f32_c16(const float* __restrict__ p) {
  v16bf r;
  cvt4(r, 0,  *(const float4*)(p + 0));
  cvt4(r, 4,  *(const float4*)(p + 4));
  cvt4(r, 8,  *(const float4*)(p + 8));
  cvt4(r, 12, *(const float4*)(p + 12));
  return r;
}

// elems 0..7 <- bf16 p[0..7]; elems 8..15 <- bf16 p[16..23]  (global or LDS)
static __device__ __forceinline__ v16bf frag_bf16_split(const __bf16* p) {
  v8bf lo = *(const v8bf*)(p);
  v8bf hi = *(const v8bf*)(p + 16);
  v16bf r;
#pragma unroll
  for (int i = 0; i < 8; ++i) { r[i] = lo[i]; r[8 + i] = hi[i]; }
  return r;
}

// elems 0..15 <- bf16 p[0..15]
static __device__ __forceinline__ v16bf frag_bf16_c16(const __bf16* p) {
  v8bf lo = *(const v8bf*)(p);
  v8bf hi = *(const v8bf*)(p + 8);
  v16bf r;
#pragma unroll
  for (int i = 0; i < 8; ++i) { r[i] = lo[i]; r[8 + i] = hi[i]; }
  return r;
}

static __device__ __forceinline__ float tanh_fast(float x) {
  float e = __expf(2.0f * x);
  return 1.0f - 2.0f / (e + 1.0f);   // exact limits at +-inf
}

// ---------------------------------------------------------------------------
// Phase 1: Y = tanh(X @ W^T + b) in bf16.
// ---------------------------------------------------------------------------
__global__ __launch_bounds__(256)
void embed_kernel(const float* __restrict__ X, const float* __restrict__ W,
                  const float* __restrict__ bias, __bf16* __restrict__ Y,
                  int transpose_out, int ng) {
  const int lane = threadIdx.x & 31;
  const int wave = threadIdx.x >> 5;
  const int lrow = lane & 15;
  const int hi   = (lane & 16) ? 1 : 0;
  const int rowB = blockIdx.x * 128 + (wave >> 1) * 32;
  const int colB = blockIdx.y * 128 + (wave & 1) * 64;

  const float* pX0 = X + (size_t)(rowB + lrow) * D_DIM + (hi ? 8 : 0);
  const float* pX1 = pX0 + 16 * D_DIM;
  const float* pW[4];
#pragma unroll
  for (int cs = 0; cs < 4; ++cs)
    pW[cs] = W + (size_t)(colB + cs * 16 + lrow) * D_DIM + (hi ? 8 : 0);

  v8f acc[2][4];
#pragma unroll
  for (int rs = 0; rs < 2; ++rs)
#pragma unroll
    for (int cs = 0; cs < 4; ++cs)
#pragma unroll
      for (int i = 0; i < 8; ++i) acc[rs][cs][i] = 0.0f;

#pragma unroll
  for (int k0 = 0; k0 < D_DIM; k0 += 32) {
    v16bf a[2];
    a[0] = frag_f32_split(pX0 + k0);
    a[1] = frag_f32_split(pX1 + k0);
#pragma unroll
    for (int cs = 0; cs < 4; ++cs) {
      v16bf bf = frag_f32_split(pW[cs] + k0);
#pragma unroll
      for (int rs = 0; rs < 2; ++rs)
        acc[rs][cs] = __builtin_amdgcn_wmma_f32_16x16x32_bf16(
            false, a[rs], false, bf, (short)0, acc[rs][cs], false, false);
    }
  }

  const int bb = transpose_out ? (rowB / ng) : 0;       // tile never straddles a batch
#pragma unroll
  for (int rs = 0; rs < 2; ++rs) {
#pragma unroll
    for (int cs = 0; cs < 4; ++cs) {
      const int c  = colB + cs * 16 + lrow;
      const float bv = bias[c];
#pragma unroll
      for (int i = 0; i < 8; ++i) {
        const int r = rowB + rs * 16 + i + hi * 8;
        const float v = tanh_fast(acc[rs][cs][i] + bv);
        if (!transpose_out) {
          Y[(size_t)r * D_DIM + c] = (__bf16)v;
        } else {
          const int m = r - bb * ng;
          Y[((size_t)bb * D_DIM + c) * ng + m] = (__bf16)v;
        }
      }
    }
  }
}

// ---------------------------------------------------------------------------
// Phase 2 (fused, flash-style), double-buffered async global->LDS staging:
// issue tile mt+1 while WMMAs consume tile mt (ASYNCcnt-tracked copies).
//   S = Lrows(16x256) @ Gtile^T  -> *1/NG -> LDS C->A relayout
//   Y += S @ Vtile ;  out = Y + local
// grid = (NL/128, B); 8 waves, wave owns 16 local rows x all 256 features.
// ---------------------------------------------------------------------------
#define SG_STRIDE 264   // 256 + 8 bf16 pad  (row = 528B, 16B aligned)
#define SV_STRIDE 72    //  64 + 8 bf16 pad  (row = 144B, 16B aligned)
#define SS_STRIDE 72
#define GBUF_B (64 * SG_STRIDE * 2)    // 33792 bytes per G buffer
#define VBUF_B (256 * SV_STRIDE * 2)   // 36864 bytes per V buffer

__global__ __launch_bounds__(256)
void attn_kernel(const __bf16* __restrict__ Lemb, const __bf16* __restrict__ Gemb,
                 const __bf16* __restrict__ Vt, const float* __restrict__ local,
                 float* __restrict__ out) {
  __shared__ __attribute__((aligned(16))) __bf16 sG[2][64 * SG_STRIDE];    // 66 KB
  __shared__ __attribute__((aligned(16))) __bf16 sV[2][256 * SV_STRIDE];   // 72 KB
  __shared__ __attribute__((aligned(16))) __bf16 sS[8][16 * SS_STRIDE];    // 18 KB

  const int tid  = threadIdx.x;
  const int lane = tid & 31;
  const int wave = tid >> 5;
  const int lrow = lane & 15;
  const int hi   = (lane & 16) ? 1 : 0;

  const size_t rowFlat = (size_t)blockIdx.y * NL_DIM + blockIdx.x * 128 + wave * 16;
  const __bf16* Gb = Gemb + (size_t)blockIdx.y * NG_DIM * D_DIM;
  const __bf16* Vb = Vt   + (size_t)blockIdx.y * D_DIM * NG_DIM;

  // per-thread async-copy bases: thread covers G row (wave+8j) col lane*16B,
  // and V row (tid/8 + 32j) col (tid%8)*16B.
  const unsigned ldsG0 = (unsigned)(size_t)(void*)sG
                         + (unsigned)(wave * (SG_STRIDE * 2) + lane * 16);
  const unsigned ldsV0 = (unsigned)(size_t)(void*)sV
                         + (unsigned)((tid >> 3) * (SV_STRIDE * 2) + (tid & 7) * 16);
  unsigned long long gG = (unsigned long long)(size_t)Gb
                          + (unsigned)(wave * 512 + lane * 16);
  unsigned long long gV = (unsigned long long)(size_t)Vb
                          + (unsigned)((tid >> 3) * (NG_DIM * 2) + (tid & 7) * 16);

  auto issue_tiles = [&](unsigned bG, unsigned bV,
                         unsigned long long ga, unsigned long long va) {
#pragma unroll
    for (int j = 0; j < 8; ++j) {
      ASYNC_B128(ldsG0 + bG + (unsigned)(j * 8 * SG_STRIDE * 2),
                 ga + (unsigned)(j * 4096));
      ASYNC_B128(ldsV0 + bV + (unsigned)(j * 32 * SV_STRIDE * 2),
                 va + (unsigned)(j * 32 * NG_DIM * 2));
    }
  };

  // --- l_emb A-fragments: loaded once, reused for all 16 m-tiles ---
  const __bf16* pL = Lemb + (rowFlat + lrow) * D_DIM + (hi ? 8 : 0);
  v16bf aL[8];
#pragma unroll
  for (int k = 0; k < 8; ++k) aL[k] = frag_bf16_split(pL + k * 32);

  v8f Y[16];
#pragma unroll
  for (int d = 0; d < 16; ++d)
#pragma unroll
    for (int i = 0; i < 8; ++i) Y[d][i] = 0.0f;

  // per-lane LDS read bases
  const __bf16* rG = &sG[0][0] + lrow * SG_STRIDE + hi * 8;
  const __bf16* rV = &sV[0][0] + lrow * SV_STRIDE + hi * 8;
  const __bf16* rS = &sS[wave][lrow * SS_STRIDE + hi * 8];
  __bf16*       wS = &sS[wave][lrow];

  // prologue: stage tile 0 into buffer 0
  issue_tiles(0u, 0u, gG, gV);
  gG += 64 * D_DIM * 2;
  gV += 64 * 2;

#pragma unroll 2                         // buffer index becomes compile-time
  for (int mt = 0; mt < 16; ++mt) {
    const int buf = mt & 1;
    WAIT_ASYNC();                        // my share of buf[mt] landed
    __syncthreads();                     // all shares landed; buf[mt^1] free

    if (mt < 15) {                       // overlap: stage tile mt+1
      issue_tiles((unsigned)((buf ^ 1) * GBUF_B), (unsigned)((buf ^ 1) * VBUF_B),
                  gG, gV);
      gG += 64 * D_DIM * 2;
      gV += 64 * 2;
    }
    const __bf16* rGb = rG + buf * (64 * SG_STRIDE);
    const __bf16* rVb = rV + buf * (256 * SV_STRIDE);

    // ---- GEMM 1: S[16x64] = Lrows @ Gtile^T (K = 256) ----
    v8f S[4];
#pragma unroll
    for (int cs = 0; cs < 4; ++cs)
#pragma unroll
      for (int i = 0; i < 8; ++i) S[cs][i] = 0.0f;

#pragma unroll
    for (int k = 0; k < 8; ++k) {
#pragma unroll
      for (int cs = 0; cs < 4; ++cs) {
        v16bf bg = frag_bf16_split(rGb + cs * 16 * SG_STRIDE + k * 32);
        S[cs] = __builtin_amdgcn_wmma_f32_16x16x32_bf16(
            false, aL[k], false, bg, (short)0, S[cs], false, false);
      }
    }

    // ---- scale by 1/NG, relayout C->A through wave-private LDS ----
#pragma unroll
    for (int cs = 0; cs < 4; ++cs)
#pragma unroll
      for (int i = 0; i < 8; ++i)
        wS[(i + hi * 8) * SS_STRIDE + cs * 16] =
            (__bf16)(S[cs][i] * (1.0f / 1024.0f));
    LDS_FENCE();                          // wave-private: in-order LDS suffices

    v16bf aS[2];
    aS[0] = frag_bf16_split(rS);
    aS[1] = frag_bf16_split(rS + 32);

    // ---- GEMM 2: Y[16x256] += S @ Vtile (K = 64) ----
#pragma unroll
    for (int d = 0; d < 16; ++d) {
#pragma unroll
      for (int k2 = 0; k2 < 2; ++k2) {
        v16bf bv = frag_bf16_split(rVb + d * 16 * SV_STRIDE + k2 * 32);
        Y[d] = __builtin_amdgcn_wmma_f32_16x16x32_bf16(
            false, aS[k2], false, bv, (short)0, Y[d], false, false);
      }
    }
  }

  // ---- epilogue: out = Y + local ----
  const size_t base = (rowFlat + hi * 8) * D_DIM + lrow;
  const float* lp = local + base;
  float*       op = out   + base;
#pragma unroll
  for (int d = 0; d < 16; ++d)
#pragma unroll
    for (int i = 0; i < 8; ++i)
      op[d * 16 + i * D_DIM] = Y[d][i] + lp[d * 16 + i * D_DIM];
}

// ---------------------------------------------------------------------------
extern "C" void kernel_launch(void* const* d_in, const int* in_sizes, int n_in,
                              void* d_out, int out_size, void* d_ws, size_t ws_size,
                              hipStream_t stream) {
  (void)in_sizes; (void)n_in; (void)out_size; (void)ws_size;
  const float* local = (const float*)d_in[0];
  const float* rawg  = (const float*)d_in[1];
  const float* Wl    = (const float*)d_in[2];
  const float* bl    = (const float*)d_in[3];
  const float* Wg    = (const float*)d_in[4];
  const float* bg    = (const float*)d_in[5];
  const float* Wv    = (const float*)d_in[6];
  const float* bv    = (const float*)d_in[7];
  float* out = (float*)d_out;

  __bf16* Lemb = (__bf16*)d_ws;                                    // 32 MB
  __bf16* Gemb = Lemb + (size_t)B_DIM * NL_DIM * D_DIM;            //  8 MB
  __bf16* Vt   = Gemb + (size_t)B_DIM * NG_DIM * D_DIM;            //  8 MB

  // Phase 1: bf16 embeddings (v_emb stored transposed per batch: [B][D][NG]).
  embed_kernel<<<dim3((B_DIM * NL_DIM) / 128, 2), 256, 0, stream>>>(
      local, Wl, bl, Lemb, 0, 1);
  embed_kernel<<<dim3((B_DIM * NG_DIM) / 128, 2), 256, 0, stream>>>(
      rawg, Wg, bg, Gemb, 0, 1);
  embed_kernel<<<dim3((B_DIM * NG_DIM) / 128, 2), 256, 0, stream>>>(
      rawg, Wv, bv, Vt, 1, NG_DIM);

  // Phase 2: fused score/value chain + residual.
  attn_kernel<<<dim3(NL_DIM / 128, B_DIM), 256, 0, stream>>>(
      Lemb, Gemb, Vt, local, out);
}
